// MultiCrossTransformer_37520834298101
// MI455X (gfx1250) — compile-verified, matrix operand order checked by reading, not run on hardware
//
#include <hip/hip_runtime.h>
#include <hip/hip_bf16.h>

// ---------------------------------------------------------------------------
// MI455X (gfx1250) implementation, all matrix math on v_wmma_f32_16x16x32_bf16:
//   (0) NCHW fp32 -> NHWC bf16 transform (makes im2col loads contiguous)
//   (1) 3x3/s2 conv as implicit GEMM   M=83200 N=512 K=4608  (~390 GFLOP)
//   (2) qk/v 1x1 projections           M=83200 N=256 K=512
//   (3) flash-attention + distance     per (b,k): [400x128]x[128x2000]x[2000x128]
// fp32 accumulation; bf16 operands.
// ---------------------------------------------------------------------------

typedef __attribute__((ext_vector_type(16))) __bf16 v16bf;
typedef __attribute__((ext_vector_type(8)))  __bf16 v8bf;
typedef __attribute__((ext_vector_type(8)))  float  v8f;

#define NIMG   208           // 8 query + 200 support images
#define HW_IN  41
#define PIX_IN 1681          // 41*41
#define OHW    20
#define MPIMG  400           // 20*20
#define CDIM   512
#define KCONV  4608          // 512*9
#define NPROJ  256           // [qk | v]

// --- 16x32 bf16 fragment (A or B operand) from an LDS tile ------------------
// Documented 16-bit layout: lane L -> row row0+(L&15); elements e<8 -> k=half*8+e,
// e>=8 -> k=16+half*8+(e-8).  => two contiguous 16-B ds loads per lane.
__device__ inline v16bf load_frag(const __bf16* base, int stride, int row0) {
  const int lane = threadIdx.x & 31;
  const int half = lane >> 4;
  const __bf16* p = base + (row0 + (lane & 15)) * stride + half * 8;
  v8bf lo = *(const v8bf*)(p);
  v8bf hi = *(const v8bf*)(p + 16);
  v16bf f;
#pragma unroll
  for (int i = 0; i < 8; ++i) { f[i] = lo[i]; f[i + 8] = hi[i]; }
  return f;
}

__device__ inline v8f wmma_bf16(v16bf a, v16bf b, v8f c) {
  return __builtin_amdgcn_wmma_f32_16x16x32_bf16(false, a, false, b, (short)0, c,
                                                 false, false);
}

// ---------------------------------------------------------------------------
// Kernel 0a: weight prep.
//  wms_bf[n*4608 + t*512 + c] = bf16(w_ms[n][c][r][s]), t=r*3+s (tap-major K:
//  every 32-wide K-block has one fixed tap and 32 consecutive channels)
//  wproj_bf[n*512 + c] = bf16(n<128 ? w_qk[n][c] : w_v[n-128][c])
// ---------------------------------------------------------------------------
__global__ void prep_weights(const float* __restrict__ w_ms,
                             const float* __restrict__ w_qk,
                             const float* __restrict__ w_v,
                             __bf16* __restrict__ wms_bf,
                             __bf16* __restrict__ wproj_bf) {
  const int N1 = CDIM * KCONV;
  const int N2 = NPROJ * CDIM;
  for (int idx = blockIdx.x * blockDim.x + threadIdx.x; idx < N1 + N2;
       idx += gridDim.x * blockDim.x) {
    if (idx < N1) {
      int o = idx / KCONV;
      int rem = idx - o * KCONV;
      int t = rem >> 9, c = rem & 511;
      wms_bf[idx] = (__bf16)w_ms[o * KCONV + c * 9 + t];
    } else {
      int j = idx - N1;
      int n = j >> 9, c = j & 511;
      float v = (n < 128) ? w_qk[n * CDIM + c] : w_v[(n - 128) * CDIM + c];
      wproj_bf[j] = (__bf16)v;
    }
  }
}

// ---------------------------------------------------------------------------
// Kernel 0b: NCHW fp32 -> NHWC bf16 (tiled transpose through LDS).
// Reads coalesced over px, writes coalesced over ch.
// ---------------------------------------------------------------------------
__global__ __launch_bounds__(256) void to_nhwc(
    const float* __restrict__ img_query, const float* __restrict__ img_supports,
    __bf16* __restrict__ nhwc) {
  const int p0  = blockIdx.x * 32;   // 53 tiles over 1681 px
  const int c0  = blockIdx.y * 64;   // 8 tiles over 512 ch
  const int img = blockIdx.z;
  const float* src = (img < 8)
      ? (img_query    + (size_t)img       * CDIM * PIX_IN)
      : (img_supports + (size_t)(img - 8) * CDIM * PIX_IN);
  __shared__ float tile[64][33];
  for (int idx = threadIdx.x; idx < 64 * 32; idx += 256) {
    int cc = idx >> 5, pp = idx & 31;
    int p = p0 + pp;
    tile[cc][pp] = (p < PIX_IN) ? src[(size_t)(c0 + cc) * PIX_IN + p] : 0.f;
  }
  __syncthreads();
  for (int idx = threadIdx.x; idx < 32 * 64; idx += 256) {
    int pp = idx >> 6, cc = idx & 63;
    int p = p0 + pp;
    if (p < PIX_IN)
      nhwc[((size_t)img * PIX_IN + p) * CDIM + c0 + cc] = (__bf16)tile[cc][pp];
  }
}

// ---------------------------------------------------------------------------
// Kernel 1: conv3x3/s2 implicit GEMM.  Block 256 thr = 8 waves (4 rowx2 col).
// Tile 64 px x 128 out-ch; 4 wmma / wave / K-step; all staging is 16-B vector.
// ---------------------------------------------------------------------------
__global__ __launch_bounds__(256) void conv_gemm(
    const __bf16* __restrict__ nhwc, const __bf16* __restrict__ wms_bf,
    __bf16* __restrict__ repr_bf) {
  const int mt  = blockIdx.x;        // 0..6   (400 = 6*64 + 16)
  const int nt  = blockIdx.y;        // 0..3
  const int img = blockIdx.z;        // 0..207
  const int m0 = mt * 64;
  const int n0 = nt * 128;

  __shared__ alignas(16) __bf16 As[64 * 32];
  __shared__ alignas(16) __bf16 Bs[128 * 32];

  const int tid  = threadIdx.x;
  const int lane = tid & 31;
  const int wave = tid >> 5;
  const int wm = wave >> 1;          // 0..3 : 16-row slice
  const int wn = wave & 1;           // 0..1 : 64-col half
  v8f acc[4] = {{}, {}, {}, {}};

  // A-staging coords: thread -> (row = tid>>2, 8-ch segment)
  const int arow = tid >> 2;
  const int aseg = (tid & 3) * 8;
  const int am   = m0 + arow;
  const int ay   = am / OHW;
  const int ax   = am - ay * OHW;
  const size_t abase = (size_t)img * PIX_IN;
  // B-staging coords: thread -> (n = tid>>1, 16-k segment)
  const int bn  = tid >> 1;
  const int bk0 = (tid & 1) << 4;
  const __bf16* brow = wms_bf + (size_t)(n0 + bn) * KCONV + bk0;

  for (int kb = 0; kb < KCONV / 32; ++kb) {   // 144 K-steps
    const int t  = kb >> 4;                   // fixed tap per K-block
    const int c0 = (kb & 15) << 5;
    const int r  = t / 3;
    const int s2 = t - r * 3;
    // --- stage A: one 16-B global load + one 16-B ds store per thread ---
    {
      v8bf v = {};
      if (am < MPIMG) {
        size_t p_in = (size_t)(ay * 2 + r) * HW_IN + (ax * 2 + s2);
        v = *(const v8bf*)(nhwc + (abase + p_in) * CDIM + c0 + aseg);
      }
      *(v8bf*)(As + arow * 32 + aseg) = v;
    }
    // --- stage B: 32 B contiguous per thread + prefetch 8 K-steps ahead ---
    {
      const __bf16* bp = brow + kb * 32;
      *(v8bf*)(Bs + bn * 32 + bk0)     = *(const v8bf*)(bp);
      *(v8bf*)(Bs + bn * 32 + bk0 + 8) = *(const v8bf*)(bp + 8);
      __builtin_prefetch(bp + 256, 0, 1);
    }
    __syncthreads();
    v16bf a = load_frag(As, 32, wm * 16);
#pragma unroll
    for (int q = 0; q < 4; ++q) {
      v16bf b = load_frag(Bs, 32, wn * 64 + q * 16);
      acc[q] = wmma_bf16(a, b, acc[q]);
    }
    __syncthreads();
  }
  // epilogue: C/D layout row = (lane>>4)*8 + j, col = lane&15
  const int rbase = wm * 16 + ((lane >> 4) << 3);
  const int col   = lane & 15;
#pragma unroll
  for (int j = 0; j < 8; ++j) {
    int m = m0 + rbase + j;
    if (m < MPIMG) {
      size_t base = ((size_t)img * MPIMG + m) * CDIM + n0 + wn * 64 + col;
#pragma unroll
      for (int q = 0; q < 4; ++q) repr_bf[base + q * 16] = (__bf16)acc[q][j];
    }
  }
}

// ---------------------------------------------------------------------------
// Kernel 2: qk/v projections.  [83200 x 512] x [512 x 256].  83200 = 1300*64,
// so no bounds checks anywhere.  Same 64x128 tiling as conv.
// ---------------------------------------------------------------------------
__global__ __launch_bounds__(256) void proj_gemm(
    const __bf16* __restrict__ repr_bf, const __bf16* __restrict__ wproj_bf,
    __bf16* __restrict__ proj_bf) {
  const int m0 = blockIdx.x * 64;    // 0..1299
  const int n0 = blockIdx.y * 128;   // 0..1

  __shared__ alignas(16) __bf16 As[64 * 32];
  __shared__ alignas(16) __bf16 Bs[128 * 32];

  const int tid  = threadIdx.x;
  const int lane = tid & 31;
  const int wave = tid >> 5;
  const int wm = wave >> 1, wn = wave & 1;
  v8f acc[4] = {{}, {}, {}, {}};

  const int arow = tid >> 2;
  const int aseg = (tid & 3) * 8;
  const __bf16* arowp = repr_bf + (size_t)(m0 + arow) * CDIM + aseg;
  const int bn  = tid >> 1;
  const int bk0 = (tid & 1) << 4;
  const __bf16* brow = wproj_bf + (size_t)(n0 + bn) * CDIM + bk0;

  for (int kb = 0; kb < CDIM / 32; ++kb) {   // 16 K-steps
    *(v8bf*)(As + arow * 32 + aseg) = *(const v8bf*)(arowp + kb * 32);
    const __bf16* bp = brow + kb * 32;
    *(v8bf*)(Bs + bn * 32 + bk0)     = *(const v8bf*)(bp);
    *(v8bf*)(Bs + bn * 32 + bk0 + 8) = *(const v8bf*)(bp + 8);
    __syncthreads();
    v16bf a = load_frag(As, 32, wm * 16);
#pragma unroll
    for (int q = 0; q < 4; ++q) {
      v16bf b = load_frag(Bs, 32, wn * 64 + q * 16);
      acc[q] = wmma_bf16(a, b, acc[q]);
    }
    __syncthreads();
  }
  const int rbase = wm * 16 + ((lane >> 4) << 3);
  const int col   = lane & 15;
#pragma unroll
  for (int j = 0; j < 8; ++j) {
    size_t base = (size_t)(m0 + rbase + j) * NPROJ + n0 + wn * 64 + col;
#pragma unroll
    for (int q = 0; q < 4; ++q) proj_bf[base + q * 16] = (__bf16)acc[q][j];
  }
}

// ---------------------------------------------------------------------------
// Kernel 3: flash attention + squared-distance partials.
// Block per (b, k, 32-row query tile).  Keys: 5 images x (6x64 + 1x16) tiles.
// ---------------------------------------------------------------------------
#define VTS 72   // padded V^T stride: rows stay 16-B aligned, banks decorrelate

__global__ __launch_bounds__(256) void attn_kernel(
    const __bf16* __restrict__ proj_bf, float* __restrict__ out) {
  const int mt   = blockIdx.x;       // 0..12
  const int kcls = blockIdx.y;       // 0..4
  const int b    = blockIdx.z;       // 0..7
  const int m0   = mt * 32;
  const float scale = 0.0883883476483184f;   // 128^-0.5
  const int qimg = b;

  __shared__ alignas(16) __bf16 Qs[32 * 128];
  __shared__ alignas(16) __bf16 Kt[64 * 128];     // [key][ch]
  __shared__ alignas(16) __bf16 Vt[128 * VTS];    // [ch][key]
  __shared__ alignas(16) __bf16 Pb[32 * 64];      // probs bf16
  __shared__ float Pf[32 * 64];                   // raw scores
  __shared__ float red[256];
  __shared__ float rowmax[32], rowsum[32], rowalpha[32];

  const int tid  = threadIdx.x;
  const int lane = tid & 31;
  const int wave = tid >> 5;
  const int rh = wave >> 2;   // row half (both phases)
  const int q4 = wave & 3;    // key quarter (sim) / channel quarter (O)

  // stage Q (16-B vector copies; zero-pad rows past 400)
  for (int idx = tid * 8; idx < 32 * 128; idx += 2048) {
    int row = idx >> 7, ch = idx & 127;
    int m = m0 + row;
    v8bf v = {};
    if (m < MPIMG)
      v = *(const v8bf*)(proj_bf + ((size_t)qimg * MPIMG + m) * NPROJ + ch);
    *(v8bf*)(Qs + idx) = v;
  }
  if (tid < 32) { rowmax[tid] = -1e30f; rowsum[tid] = 0.f; }
  v8f o0 = {}, o1 = {};
  __syncthreads();

  for (int n = 0; n < 5; ++n) {
    const size_t sbase = (size_t)(8 + (b * 5 + kcls) * 5 + n) * MPIMG;
    for (int kt = 0; kt < 7; ++kt) {
      const int key0  = kt * 64;
      const int valid = (kt < 6) ? 64 : 16;
      // stage K tile [key][ch] (vector copies)
      for (int idx = tid * 8; idx < valid * 128; idx += 2048) {
        int kk = idx >> 7, ch = idx & 127;
        *(v8bf*)(Kt + kk * 128 + ch) =
            *(const v8bf*)(proj_bf + (sbase + key0 + kk) * NPROJ + ch);
      }
      // stage V^T [ch][key]: vector global read, scalar LDS transpose writes
      for (int idx = tid * 8; idx < valid * 128; idx += 2048) {
        int kk = idx >> 7, ch = idx & 127;
        v8bf v = *(const v8bf*)(proj_bf + (sbase + key0 + kk) * NPROJ + 128 + ch);
#pragma unroll
        for (int i = 0; i < 8; ++i) Vt[(ch + i) * VTS + kk] = v[i];
      }
      __syncthreads();
      // sim tile: rows rh*16.., keys q4*16.. ; K over 128 channels
      v8f s = {};
#pragma unroll
      for (int kc = 0; kc < 4; ++kc) {
        v16bf a  = load_frag(Qs + kc * 32, 128, rh * 16);
        v16bf bb = load_frag(Kt + kc * 32, 128, q4 * 16);
        s = wmma_bf16(a, bb, s);
      }
      {
        int rb  = rh * 16 + ((lane >> 4) << 3);
        int cc  = q4 * 16 + (lane & 15);
#pragma unroll
        for (int j = 0; j < 8; ++j) Pf[(rb + j) * 64 + cc] = s[j] * scale;
      }
      __syncthreads();
      // --- online softmax, 8 threads per row + 32-thread reduction ---
      {
        const int r = tid >> 3, seg = (tid & 7) * 8;
        float lm = -1e30f;
#pragma unroll
        for (int i = 0; i < 8; ++i) {
          int c = seg + i;
          if (c < valid) lm = fmaxf(lm, Pf[r * 64 + c]);
        }
        red[tid] = lm;
      }
      __syncthreads();
      if (tid < 32) {
        float mold = rowmax[tid], mnew = mold;
#pragma unroll
        for (int i = 0; i < 8; ++i) mnew = fmaxf(mnew, red[tid * 8 + i]);
        rowalpha[tid] = __expf(mold - mnew);
        rowmax[tid]   = mnew;
      }
      __syncthreads();
      {
        const int r = tid >> 3, seg = (tid & 7) * 8;
        float mnew = rowmax[r];
        float ls = 0.f;
#pragma unroll
        for (int i = 0; i < 8; ++i) {
          int c = seg + i;
          float p = (c < valid) ? __expf(Pf[r * 64 + c] - mnew) : 0.f;
          Pb[r * 64 + c] = (__bf16)p;
          ls += p;
        }
        red[tid] = ls;
      }
      __syncthreads();
      if (tid < 32) {
        float ssum = rowsum[tid] * rowalpha[tid];
#pragma unroll
        for (int i = 0; i < 8; ++i) ssum += red[tid * 8 + i];
        rowsum[tid] = ssum;
      }
      __syncthreads();
      // rescale accumulators, then O += P x V (K over 64 keys)
      {
        int rb = rh * 16 + ((lane >> 4) << 3);
#pragma unroll
        for (int j = 0; j < 8; ++j) {
          float al = rowalpha[rb + j];
          o0[j] *= al; o1[j] *= al;
        }
      }
#pragma unroll
      for (int kc = 0; kc < 2; ++kc) {
        v16bf a  = load_frag(Pb + kc * 32, 64, rh * 16);
        v16bf b0 = load_frag(Vt + kc * 32, VTS, q4 * 32);
        v16bf b1 = load_frag(Vt + kc * 32, VTS, q4 * 32 + 16);
        o0 = wmma_bf16(a, b0, o0);
        o1 = wmma_bf16(a, b1, o1);
      }
      __syncthreads();
    }
  }
  // finalize: normalize O, squared distance vs query_v, one f32 atomic per lane
  float part = 0.f;
  {
    int rb = rh * 16 + ((lane >> 4) << 3);
    int cc = q4 * 32 + (lane & 15);
#pragma unroll
    for (int j = 0; j < 8; ++j) {
      int m = m0 + rb + j;
      if (m < MPIMG) {
        float inv = 1.f / rowsum[rb + j];
        size_t qb = ((size_t)qimg * MPIMG + m) * NPROJ + 128;
        float d0 = (float)proj_bf[qb + cc]      - o0[j] * inv;
        float d1 = (float)proj_bf[qb + cc + 16] - o1[j] * inv;
        part += d0 * d0 + d1 * d1;
      }
    }
  }
  atomicAdd(&out[b * 5 + kcls], -part * (1.f / 400.f));
}

__global__ void zero_out(float* out, int n) {
  int i = blockIdx.x * blockDim.x + threadIdx.x;
  if (i < n) out[i] = 0.f;
}

// ---------------------------------------------------------------------------
extern "C" void kernel_launch(void* const* d_in, const int* in_sizes, int n_in,
                              void* d_out, int out_size, void* d_ws, size_t ws_size,
                              hipStream_t stream) {
  (void)in_sizes; (void)n_in; (void)ws_size;
  const float* img_query    = (const float*)d_in[0];
  const float* img_supports = (const float*)d_in[1];
  const float* w_qk         = (const float*)d_in[2];
  const float* w_v          = (const float*)d_in[3];
  const float* w_ms         = (const float*)d_in[4];
  float* out = (float*)d_out;

  // workspace carve (bf16): conv W 4.7MB | proj W 0.26MB | repr 85MB | nhwc 358MB
  // proj output (43MB) aliases the nhwc buffer (dead after conv_gemm).
  __bf16* wms_bf   = (__bf16*)d_ws;
  __bf16* wproj_bf = wms_bf   + (size_t)CDIM * KCONV;
  __bf16* repr_bf  = wproj_bf + (size_t)NPROJ * CDIM;
  __bf16* nhwc     = repr_bf  + (size_t)NIMG * MPIMG * CDIM;
  __bf16* proj_bf  = nhwc;   // alias: proj_gemm only reads repr_bf

  prep_weights<<<2048, 256, 0, stream>>>(w_ms, w_qk, w_v, wms_bf, wproj_bf);
  to_nhwc<<<dim3(53, 8, NIMG), 256, 0, stream>>>(img_query, img_supports, nhwc);
  conv_gemm<<<dim3(7, 4, NIMG), 256, 0, stream>>>(nhwc, wms_bf, repr_bf);
  proj_gemm<<<dim3(1300, 2), 256, 0, stream>>>(repr_bf, wproj_bf, proj_bf);
  zero_out<<<1, 64, 0, stream>>>(out, out_size);
  attn_kernel<<<dim3(13, 5, 8), 256, 0, stream>>>(proj_bf, out);
}
